// MambaBlock_76596446756870
// MI455X (gfx1250) — matrix operand order repaired and checked by reading, not compiled, not run on hardware
//
#include <hip/hip_runtime.h>

// ---------------------------------------------------------------------------
// Problem constants (from reference)
// ---------------------------------------------------------------------------
#define D_MODEL 1024
#define D_STATE 16
#define D_CONV  4
#define D_INNER 2048
#define DT_RANK 128
#define BATCH   2
#define SEQLEN  1024
#define MTOT    (BATCH * SEQLEN)             // 2048 rows for all GEMMs
#define XPROJ_N (DT_RANK + 2 * D_STATE)      // 160

// ---------------------------------------------------------------------------
// CDNA5 WMMA types (wave32): 16x16x32 bf16 -> f32 accumulate
// ---------------------------------------------------------------------------
typedef __attribute__((ext_vector_type(16))) __bf16 v16bf;
typedef __attribute__((ext_vector_type(8)))  float  v8f;

__device__ __forceinline__ __bf16 f2bf(float f) {
    // round-to-nearest-even fp32 -> bf16 via integer path
    unsigned int u = __builtin_bit_cast(unsigned int, f);
    u += 0x7FFFu + ((u >> 16) & 1u);
    unsigned short s = (unsigned short)(u >> 16);
    return __builtin_bit_cast(__bf16, s);
}

__device__ __forceinline__ float sigmoid_f(float x) { return 1.f / (1.f + __expf(-x)); }
__device__ __forceinline__ float silu_f(float x)    { return x * sigmoid_f(x); }
__device__ __forceinline__ float softplus_f(float x) {
    return (x > 20.f) ? x : __logf(1.f + __expf(x));
}

// ---------------------------------------------------------------------------
// Fragment packing (fp32 -> bf16, WMMA per-lane layout, ISA 7.12.2)
//
// A (16x32 bf16 tile): lane = half*16 + (row&15); element e:
//   e<8  -> K = half*8 + e ; e>=8 -> K = 16 + half*8 + (e-8)
// B (32x16 bf16 tile): lane = half*16 + (col&15); element e: same K map as A
// Packed offset: (((tile * nkb + kb) * 32 + lane) * 16 + e) -> each lane's
// fragment is 32 contiguous bytes (one 16x32 A slab = 1024 B).
// ---------------------------------------------------------------------------
__global__ void pack_a_kernel(const float* __restrict__ src, int ld, int col0,
                              __bf16* __restrict__ dst, int M, int K) {
    size_t i = (size_t)blockIdx.x * blockDim.x + threadIdx.x;
    if (i >= (size_t)M * K) return;
    const int r = (int)(i / K);
    const int k = (int)(i % K);
    const int tm = r >> 4, rl = r & 15;
    const int kb = k >> 5, kr = k & 31;
    int h, e;
    if (kr < 16) { h = kr >> 3; e = kr & 7; }
    else         { h = (kr - 16) >> 3; e = 8 + ((kr - 16) & 7); }
    const int lane = h * 16 + rl;
    const size_t o = (((size_t)tm * (K >> 5) + kb) * 32 + lane) * 16 + e;
    dst[o] = f2bf(src[(size_t)r * ld + col0 + k]);
}

__global__ void pack_b_kernel(const float* __restrict__ src,
                              __bf16* __restrict__ dst, int K, int N) {
    size_t i = (size_t)blockIdx.x * blockDim.x + threadIdx.x;
    if (i >= (size_t)K * N) return;
    const int k = (int)(i / N);
    const int c = (int)(i % N);
    const int tn = c >> 4, cl = c & 15;
    const int kb = k >> 5, kr = k & 31;
    const int h = (kr & 15) >> 3, e = (kr & 7) + ((kr >> 4) << 3);
    const int lane = h * 16 + cl;
    const size_t o = (((size_t)tn * (K >> 5) + kb) * 32 + lane) * 16 + e;
    dst[o] = f2bf(src[i]);
}

// ---------------------------------------------------------------------------
// Staged GEMM: C[MxN] = A[MxK] * B[KxN], packed bf16 fragments.
// Grid: (x = n-tile-groups/8, y = m-tiles). All 8 waves of a block share one
// 16-row A strip, staged into LDS once via global_load_async_to_lds_b128
// (ASYNCcnt) and read back with ds_load_b128. Fragments are ping-pong
// double-buffered (unroll-by-2) so loads land directly in the WMMA operand
// registers -- no rotation copies -- while the previous slab multiplies.
// Requires: (N/16) % NT == 0, (N/16/NT) % 8 == 0, K % 64 == 0 or K==128 etc.
// (nkb = K/32 must be even), K*32 <= 64KB.
// ---------------------------------------------------------------------------
template<int NT>
__global__ void __launch_bounds__(256)
gemm_bf16_wmma_smem_kernel(const __bf16* __restrict__ Ap,
                           const __bf16* __restrict__ Bp,
                           float* __restrict__ C, int M, int N, int K) {
    extern __shared__ __align__(128) __bf16 As[];   // K*32 bytes (A strip)
    const int lane = threadIdx.x & 31;
    const int wave = threadIdx.x >> 5;
    const int nkb  = K >> 5;                        // even at every call site
    const int tm   = blockIdx.y;
    const int tg   = blockIdx.x * 8 + wave;

    // ---- async copy of the packed A strip (nkb * 1024 B) into LDS ----------
    {
        const char* gA = (const char*)(Ap + (size_t)tm * nkb * 512);
        const unsigned lbase = (unsigned)(uintptr_t)(void*)&As[0];
        const int chunks = nkb * 64;                // 16-byte chunks
        for (int c = threadIdx.x; c < chunks; c += 256) {
            const unsigned loff = lbase + (unsigned)c * 16u;
            const char* ga = gA + (size_t)c * 16;
            asm volatile("global_load_async_to_lds_b128 %0, %1, off"
                         :: "v"(loff), "v"(ga) : "memory");
        }
        asm volatile("s_wait_asynccnt 0" ::: "memory");
        __syncthreads();
    }

    const __bf16* __restrict__ Bb[NT];
#pragma unroll
    for (int j = 0; j < NT; ++j)
        Bb[j] = Bp + ((size_t)(tg * NT + j) * nkb * 32 + lane) * 16;
    const __bf16* Al = &As[lane * 16];

    v8f acc[NT];
#pragma unroll
    for (int j = 0; j < NT; ++j) acc[j] = (v8f)0.f;

    // ---- ping-pong buffers: set 0 holds even slabs, set 1 odd slabs --------
    v16bf a0 = *(const v16bf*)Al;                   // ds_load (slab 0)
    v16bf b0[NT];
#pragma unroll
    for (int j = 0; j < NT; ++j) b0[j] = *(const v16bf*)Bb[j];

    for (int kb = 0; kb < nkb; kb += 2) {
        // fill set 1 with slab kb+1 (always exists: nkb even)
        const v16bf a1 = *(const v16bf*)(Al + (size_t)(kb + 1) * 512);
        v16bf b1[NT];
#pragma unroll
        for (int j = 0; j < NT; ++j)
            b1[j] = *(const v16bf*)(Bb[j] + (size_t)(kb + 1) * 512);
        // multiply slab kb (set 0) while set-1 loads are in flight
#pragma unroll
        for (int j = 0; j < NT; ++j)
            acc[j] = __builtin_amdgcn_wmma_f32_16x16x32_bf16(
                false, a0, false, b0[j], (short)0, acc[j], false, false);
        // refill set 0 with slab kb+2 (loads define the loop-carried regs)
        if (kb + 2 < nkb) {
            a0 = *(const v16bf*)(Al + (size_t)(kb + 2) * 512);
#pragma unroll
            for (int j = 0; j < NT; ++j)
                b0[j] = *(const v16bf*)(Bb[j] + (size_t)(kb + 2) * 512);
        }
        // multiply slab kb+1 (set 1)
#pragma unroll
        for (int j = 0; j < NT; ++j)
            acc[j] = __builtin_amdgcn_wmma_f32_16x16x32_bf16(
                false, a1, false, b1[j], (short)0, acc[j], false, false);
    }

    // ---- store: C/D f32 16x16: VGPR v -> M = v + 8*half; N = lane&15 --------
    const int row = lane & 15, half = lane >> 4;
#pragma unroll
    for (int j = 0; j < NT; ++j) {
        float* __restrict__ crow =
            C + (size_t)(tm * 16 + half * 8) * N + ((tg * NT + j) * 16 + row);
#pragma unroll
        for (int v = 0; v < 8; ++v) crow[(size_t)v * N] = acc[j][v];
    }
}

// ---------------------------------------------------------------------------
// Global-only pipelined GEMM (for shapes whose n-tile groups don't divide by
// blocks, e.g. N=160). One wave per 16x(16*NT) strip, 1D grid with guard.
// Same ping-pong structure; nkb must be even.
// ---------------------------------------------------------------------------
template<int NT>
__global__ void __launch_bounds__(256)
gemm_bf16_wmma_kernel(const __bf16* __restrict__ Ap, const __bf16* __restrict__ Bp,
                      float* __restrict__ C, int M, int N, int K) {
    const int lane = threadIdx.x & 31;
    const int wave = threadIdx.x >> 5;
    const int ntn  = N >> 4;
    const int ngrp = ntn / NT;
    const int groups = (M >> 4) * ngrp;
    const int t = blockIdx.x * 8 + wave;            // wave-uniform tile id
    if (t >= groups) return;                         // EXEC stays all-1s
    const int tm = t / ngrp;
    const int tg = t - tm * ngrp;
    const int nkb = K >> 5;

    const __bf16* __restrict__ Ab = Ap + ((size_t)tm * nkb * 32 + lane) * 16;
    const __bf16* __restrict__ Bb[NT];
#pragma unroll
    for (int j = 0; j < NT; ++j)
        Bb[j] = Bp + ((size_t)(tg * NT + j) * nkb * 32 + lane) * 16;

    v8f acc[NT];
#pragma unroll
    for (int j = 0; j < NT; ++j) acc[j] = (v8f)0.f;

    v16bf a0 = *(const v16bf*)Ab;
    v16bf b0[NT];
#pragma unroll
    for (int j = 0; j < NT; ++j) b0[j] = *(const v16bf*)Bb[j];

    for (int kb = 0; kb < nkb; kb += 2) {
        const v16bf a1 = *(const v16bf*)(Ab + (size_t)(kb + 1) * 512);
        v16bf b1[NT];
#pragma unroll
        for (int j = 0; j < NT; ++j)
            b1[j] = *(const v16bf*)(Bb[j] + (size_t)(kb + 1) * 512);
        if (kb + 2 < nkb)                            // hint the slab after next
            __builtin_prefetch(Ab + (size_t)(kb + 2) * 512, 0, 3);
#pragma unroll
        for (int j = 0; j < NT; ++j)
            acc[j] = __builtin_amdgcn_wmma_f32_16x16x32_bf16(
                false, a0, false, b0[j], (short)0, acc[j], false, false);
        if (kb + 2 < nkb) {
            a0 = *(const v16bf*)(Ab + (size_t)(kb + 2) * 512);
#pragma unroll
            for (int j = 0; j < NT; ++j)
                b0[j] = *(const v16bf*)(Bb[j] + (size_t)(kb + 2) * 512);
        }
#pragma unroll
        for (int j = 0; j < NT; ++j)
            acc[j] = __builtin_amdgcn_wmma_f32_16x16x32_bf16(
                false, a1, false, b1[j], (short)0, acc[j], false, false);
    }

    const int row = lane & 15, half = lane >> 4;
#pragma unroll
    for (int j = 0; j < NT; ++j) {
        float* __restrict__ crow =
            C + (size_t)(tm * 16 + half * 8) * N + ((tg * NT + j) * 16 + row);
#pragma unroll
        for (int v = 0; v < 8; ++v) crow[(size_t)v * N] = acc[j][v];
    }
}

// ---------------------------------------------------------------------------
// Depthwise causal conv1d (width 4) + bias + SiLU.
// x_in is columns [0, D_INNER) of xz (row stride 2*D_INNER).
// ---------------------------------------------------------------------------
__global__ void conv_silu_kernel(const float* __restrict__ xz,
                                 const float* __restrict__ ck,   // (D_CONV,1,D_INNER)
                                 const float* __restrict__ cb,   // (D_INNER)
                                 float* __restrict__ u) {
    size_t i = (size_t)blockIdx.x * blockDim.x + threadIdx.x;
    if (i >= (size_t)MTOT * D_INNER) return;
    const int d = (int)(i % D_INNER);
    const int m = (int)(i / D_INNER);
    const int l = m % SEQLEN;
    float s = cb[d];
#pragma unroll
    for (int w = 0; w < D_CONV; ++w) {
        const int ls = l + w - (D_CONV - 1);         // causal left pad
        if (ls >= 0)
            s += ck[w * D_INNER + d] *
                 xz[(size_t)(m + w - (D_CONV - 1)) * (2 * D_INNER) + d];
    }
    u[i] = silu_f(s);
}

// dt = softplus(dt_pre + b_dt)
__global__ void dt_softplus_kernel(const float* __restrict__ pre,
                                   const float* __restrict__ bdt,
                                   float* __restrict__ dt) {
    size_t i = (size_t)blockIdx.x * blockDim.x + threadIdx.x;
    if (i >= (size_t)MTOT * D_INNER) return;
    dt[i] = softplus_f(pre[i] + bdt[i % D_INNER]);
}

// ---------------------------------------------------------------------------
// Selective scan: one thread per (b, d) channel, h[16] held in VGPRs,
// per-timestep B/C vectors staged once per block through LDS.
// ---------------------------------------------------------------------------
__global__ void __launch_bounds__(256)
scan_kernel(const float* __restrict__ dt, const float* __restrict__ u,
            const float* __restrict__ xdbl, const float* __restrict__ A_log,
            const float* __restrict__ Dcoef, float* __restrict__ y) {
    __shared__ float sBC[2 * D_STATE];
    const int blocks_per_b = D_INNER / 256;
    const int b = blockIdx.x / blocks_per_b;
    const int d = (blockIdx.x % blocks_per_b) * 256 + threadIdx.x;

    float Areg[D_STATE], h[D_STATE];
#pragma unroll
    for (int n = 0; n < D_STATE; ++n) {
        Areg[n] = -__expf(A_log[(size_t)d * D_STATE + n]);
        h[n] = 0.f;
    }
    const float Dd = Dcoef[d];

    for (int l = 0; l < SEQLEN; ++l) {
        const size_t m = (size_t)b * SEQLEN + l;
        if (threadIdx.x < 2 * D_STATE)
            sBC[threadIdx.x] = xdbl[m * XPROJ_N + DT_RANK + threadIdx.x];
        __syncthreads();

        const float dtv = dt[m * D_INNER + d];
        const float uv  = u[m * D_INNER + d];
        float acc = 0.f;
#pragma unroll
        for (int n = 0; n < D_STATE; ++n) {
            const float dA = __expf(dtv * Areg[n]);
            h[n] = dA * h[n] + (dtv * sBC[n]) * uv;
            acc += h[n] * sBC[D_STATE + n];
        }
        y[m * D_INNER + d] = acc + uv * Dd;          // skip connection u*D
        __syncthreads();
    }
}

// g = y * silu(z), emitted directly as bf16 in packed-A fragment layout
__global__ void gate_pack_kernel(const float* __restrict__ y,
                                 const float* __restrict__ xz,
                                 __bf16* __restrict__ gp) {
    size_t i = (size_t)blockIdx.x * blockDim.x + threadIdx.x;
    if (i >= (size_t)MTOT * D_INNER) return;
    const int d = (int)(i % D_INNER);
    const size_t m = i / D_INNER;
    const float z = xz[m * (2 * D_INNER) + D_INNER + d];
    const float g = y[i] * silu_f(z);
    const int tm = (int)(m >> 4), rl = (int)(m & 15);
    const int kb = d >> 5, kr = d & 31;
    int h, e;
    if (kr < 16) { h = kr >> 3; e = kr & 7; }
    else         { h = (kr - 16) >> 3; e = 8 + ((kr - 16) & 7); }
    const int lane = h * 16 + rl;
    const size_t o = (((size_t)tm * (D_INNER >> 5) + kb) * 32 + lane) * 16 + e;
    gp[o] = f2bf(g);
}

// ---------------------------------------------------------------------------
// Launch
// ---------------------------------------------------------------------------
extern "C" void kernel_launch(void* const* d_in, const int* in_sizes, int n_in,
                              void* d_out, int out_size, void* d_ws, size_t ws_size,
                              hipStream_t stream) {
    const float* x      = (const float*)d_in[0];   // (B, L, D_MODEL)
    const float* W_in   = (const float*)d_in[1];   // (D_MODEL, 2*D_INNER)
    const float* conv_k = (const float*)d_in[2];   // (D_CONV, 1, D_INNER)
    const float* conv_b = (const float*)d_in[3];   // (D_INNER)
    const float* W_xprj = (const float*)d_in[4];   // (D_INNER, XPROJ_N)
    const float* W_dt   = (const float*)d_in[5];   // (DT_RANK, D_INNER)
    const float* b_dt   = (const float*)d_in[6];   // (D_INNER)
    const float* A_log  = (const float*)d_in[7];   // (D_INNER, D_STATE)
    const float* Dco    = (const float*)d_in[8];   // (D_INNER)
    const float* W_out  = (const float*)d_in[9];   // (D_INNER, D_MODEL)
    float* out = (float*)d_out;

    // ---- workspace carve-out ------------------------------------------------
    char* ws = (char*)d_ws;
    size_t off = 0;
    auto alloc = [&](size_t bytes) -> void* {
        void* p = ws + off;
        off += (bytes + 255) & ~(size_t)255;
        return p;
    };
    float*  xz     = (float*)alloc((size_t)MTOT * 2 * D_INNER * 4);
    float*  u      = (float*)alloc((size_t)MTOT * D_INNER * 4);
    float*  xdbl   = (float*)alloc((size_t)MTOT * XPROJ_N * 4);
    float*  dtpre  = (float*)alloc((size_t)MTOT * D_INNER * 4);
    float*  dt     = (float*)alloc((size_t)MTOT * D_INNER * 4);
    float*  y      = (float*)alloc((size_t)MTOT * D_INNER * 4);
    __bf16* x_p    = (__bf16*)alloc((size_t)MTOT * D_MODEL * 2);
    __bf16* Win_p  = (__bf16*)alloc((size_t)D_MODEL * 2 * D_INNER * 2);
    __bf16* u_p    = (__bf16*)alloc((size_t)MTOT * D_INNER * 2);
    __bf16* Wxp_p  = (__bf16*)alloc((size_t)D_INNER * XPROJ_N * 2);
    __bf16* dtl_p  = (__bf16*)alloc((size_t)MTOT * DT_RANK * 2);
    __bf16* Wdt_p  = (__bf16*)alloc((size_t)DT_RANK * D_INNER * 2);
    __bf16* g_p    = (__bf16*)alloc((size_t)MTOT * D_INNER * 2);
    __bf16* Wout_p = (__bf16*)alloc((size_t)D_INNER * D_MODEL * 2);

    auto blocks = [](size_t n) { return (unsigned)((n + 255) / 256); };

    // ---- stage 1: xz = x @ W_in (LDS-staged WMMA) ---------------------------
    pack_a_kernel<<<blocks((size_t)MTOT * D_MODEL), 256, 0, stream>>>(
        x, D_MODEL, 0, x_p, MTOT, D_MODEL);
    pack_b_kernel<<<blocks((size_t)D_MODEL * 2 * D_INNER), 256, 0, stream>>>(
        W_in, Win_p, D_MODEL, 2 * D_INNER);
    {
        dim3 g((2 * D_INNER / 16 / 4) / 8, MTOT / 16);     // (8, 128)
        gemm_bf16_wmma_smem_kernel<4><<<g, 256, D_MODEL * 32, stream>>>(
            x_p, Win_p, xz, MTOT, 2 * D_INNER, D_MODEL);
    }

    // ---- stage 2: causal depthwise conv + SiLU -> u -------------------------
    conv_silu_kernel<<<blocks((size_t)MTOT * D_INNER), 256, 0, stream>>>(
        xz, conv_k, conv_b, u);

    // ---- stage 3: x_dbl = u @ W_xproj (N=160 -> global-only NT=2) -----------
    pack_a_kernel<<<blocks((size_t)MTOT * D_INNER), 256, 0, stream>>>(
        u, D_INNER, 0, u_p, MTOT, D_INNER);
    pack_b_kernel<<<blocks((size_t)D_INNER * XPROJ_N), 256, 0, stream>>>(
        W_xprj, Wxp_p, D_INNER, XPROJ_N);
    {
        unsigned groups = (MTOT / 16) * (XPROJ_N / 16 / 2); // 128*5 = 640
        gemm_bf16_wmma_kernel<2><<<(groups + 7) / 8, 256, 0, stream>>>(
            u_p, Wxp_p, xdbl, MTOT, XPROJ_N, D_INNER);
    }

    // ---- stage 4: dt = softplus(dt_low @ W_dt + b_dt) -----------------------
    pack_a_kernel<<<blocks((size_t)MTOT * DT_RANK), 256, 0, stream>>>(
        xdbl, XPROJ_N, 0, dtl_p, MTOT, DT_RANK);
    pack_b_kernel<<<blocks((size_t)DT_RANK * D_INNER), 256, 0, stream>>>(
        W_dt, Wdt_p, DT_RANK, D_INNER);
    {
        dim3 g((D_INNER / 16 / 4) / 8, MTOT / 16);          // (4, 128)
        gemm_bf16_wmma_smem_kernel<4><<<g, 256, DT_RANK * 32, stream>>>(
            dtl_p, Wdt_p, dtpre, MTOT, D_INNER, DT_RANK);
    }
    dt_softplus_kernel<<<blocks((size_t)MTOT * D_INNER), 256, 0, stream>>>(
        dtpre, b_dt, dt);

    // ---- stage 5: selective scan (register-resident h, LDS-shared B/C) ------
    scan_kernel<<<BATCH * (D_INNER / 256), 256, 0, stream>>>(
        dt, u, xdbl, A_log, Dco, y);

    // ---- stage 6: out = (y * silu(z)) @ W_out -------------------------------
    gate_pack_kernel<<<blocks((size_t)MTOT * D_INNER), 256, 0, stream>>>(y, xz, g_p);
    pack_b_kernel<<<blocks((size_t)D_INNER * D_MODEL), 256, 0, stream>>>(
        W_out, Wout_p, D_INNER, D_MODEL);
    {
        dim3 g((D_MODEL / 16 / 4) / 8, MTOT / 16);          // (2, 128)
        gemm_bf16_wmma_smem_kernel<4><<<g, 256, D_INNER * 32, stream>>>(
            g_p, Wout_p, out, MTOT, D_MODEL, D_INNER);
    }
}
